// MixEHR_5428838662489
// MI455X (gfx1250) — matrix operand order repaired and checked by reading, not compiled, not run on hardware
//
#include <hip/hip_runtime.h>

// MixEHR SCVB0 step fused for MI455X (gfx1250, wave32, WMMA bf16).
//
// out[V,K] = (1-rho)*exp_n + rho*scale * b * ( (BOW/(a@b.T + eps)).T @ a )
//   a = alpha + exp_m[batch_indices]   [B=512, K=50]  -> bf16, padded K'=64
//   b = (beta+exp_n)/denom             [V=100000, 50] -> bf16 on the fly
//
// Memory-bound target: ~265 MB @ 23.3 TB/s. Both GEMMs run on
// v_wmma_f32_16x16x32_bf16 with f32 accumulation; Z and W live only in
// registers (the WMMA D layout of GEMM1 is bit-for-bit the A layout of GEMM2).
// All divides use v_rcp_f32 (results are truncated to bf16 anyway) so the
// hot loop stays load/WMMA-bound instead of div-sequence VALU-bound.

typedef __bf16 bf16_t;
typedef __attribute__((ext_vector_type(16))) __bf16 v16bf;
typedef __attribute__((ext_vector_type(8)))  __bf16 v8bf;
typedef __attribute__((ext_vector_type(8)))  float  v8f;

#define VV 100000
#define KK 50
#define KP 64
#define BB 512
#define NSTRIP (VV / 16)   // 6250 16-row V strips

union V16U { v16bf v; v8bf h[2]; };

__device__ __forceinline__ float fastrcp(float x) {
    return __builtin_amdgcn_rcpf(x);
}

// ---------------------------------------------------------------- denom[K]
__global__ __launch_bounds__(256)
void mixehr_denom_kernel(const float* __restrict__ beta,
                         const float* __restrict__ expn,
                         float* __restrict__ denom) {
    __shared__ float part[64];
    if (threadIdx.x < 64) part[threadIdx.x] = 0.0f;
    __syncthreads();
    const int N = VV * KK;                       // 5,000,000
    for (int i = blockIdx.x * blockDim.x + threadIdx.x; i < N;
         i += gridDim.x * blockDim.x) {
        int k = i - (i / KK) * KK;               // i % 50 (distinct per lane)
        atomicAdd(&part[k], beta[i] + expn[i]);
    }
    __syncthreads();
    if (threadIdx.x < KK) atomicAdd(&denom[threadIdx.x], part[threadIdx.x]);
}

// ------------------------------------------- a (row-major + transposed) bf16
__global__ __launch_bounds__(256)
void mixehr_prep_a_kernel(const float* __restrict__ alpha,
                          const float* __restrict__ expm,
                          const int*   __restrict__ bidx,
                          bf16_t* __restrict__ aR,    // [512][64]
                          bf16_t* __restrict__ aT) {  // [64][512]
    int t = blockIdx.x * blockDim.x + threadIdx.x;    // 32768 threads
    int d = t >> 6;
    int k = t & 63;
    float val = 0.0f;
    if (k < KK) val = alpha[k] + expm[(long)bidx[d] * KK + k];
    bf16_t h = (bf16_t)val;
    aR[d * KP + k] = h;
    aT[k * BB + d] = h;
}

// --------------------------------------------------------------- main fused
__global__ __launch_bounds__(256)
void mixehr_main_kernel(const float*  __restrict__ BOW,    // [512][100000]
                        const float*  __restrict__ beta,   // [V][50]
                        const float*  __restrict__ expn,   // [V][50]
                        const float*  __restrict__ denom,  // [64] (>=50 valid)
                        const bf16_t* __restrict__ aR,     // [512][64]
                        const bf16_t* __restrict__ aT,     // [64][512]
                        const int*       __restrict__ iter_n,
                        const long long* __restrict__ batch_C,
                        const long long* __restrict__ C_m,
                        float* __restrict__ out) {         // [V][50]
    const int lane  = threadIdx.x & 31;
    const int wave  = threadIdx.x >> 5;
    const int strip = blockIdx.x * 8 + wave;
    if (strip >= NSTRIP) return;

    const int v0    = strip * 16;
    const int lhalf = lane >> 4;        // 0 | 1
    const int lmod  = lane & 15;
    const int vrow  = v0 + lmod;        // this lane's v for B-operand / GEMM2 A rows

    const float rho   = __powf((float)(*iter_n) + 5.0f, -0.9f);
    const float scale = (float)((double)(*C_m) / (double)(*batch_C));
    const float rs    = rho * scale;

    // ---- GEMM1 B operand: b[v][k'] = (beta+exp_n)*rcp(denom), bf16, padded 64
    // lane layout (16-bit B, 32x16): col n = lane%16, K = chunk*32 + lhalf*16 + j
    v16bf Bop0, Bop1;
#pragma unroll
    for (int j = 0; j < 16; ++j) {
        int k0 = lhalf * 16 + j;            // 0..31  (always < 50)
        int k1 = 32 + lhalf * 16 + j;       // 32..63 (guard)
        long base = (long)vrow * KK;
        float b0 = (beta[base + k0] + expn[base + k0]) * fastrcp(denom[k0]);
        float b1 = (k1 < KK)
                 ? (beta[base + k1] + expn[base + k1]) * fastrcp(denom[k1])
                 : 0.0f;
        Bop0[j] = (bf16_t)b0;
        Bop1[j] = (bf16_t)b1;
    }

    // temp accumulator: 4 n-tiles of k' (16 each), D layout v8f per tile
    v8f acc[4];
#pragma unroll
    for (int t = 0; t < 4; ++t)
#pragma unroll
        for (int r = 0; r < 8; ++r) acc[t][r] = 0.0f;

    // ---- contraction over 512 docs in 16 blocks of 32
    for (int db = 0; db < 16; ++db) {
        v16bf Wop;                           // GEMM2 A operand (16v x 32doc) bf16
#pragma unroll
        for (int sub = 0; sub < 2; ++sub) {
            const int  d0    = db * 32 + sub * 16;
            const long abase = (long)(d0 + lmod) * KP;  // A row = doc = d0+lane%16

            // 16-bit A layout (16x32): per lane 16 contiguous K' at offset
            // lhalf*16 (then +32 for the second K' chunk), matching Bop
            V16U A0, A1;
            A0.h[0] = *(const v8bf*)(aR + abase + lhalf * 16);
            A0.h[1] = *(const v8bf*)(aR + abase + lhalf * 16 + 8);
            A1.h[0] = *(const v8bf*)(aR + abase + 32 + lhalf * 16);
            A1.h[1] = *(const v8bf*)(aR + abase + 32 + lhalf * 16 + 8);

            // Z tile [16 doc x 16 v] over K'=64
            v8f z = {0.f, 0.f, 0.f, 0.f, 0.f, 0.f, 0.f, 0.f};
            z = __builtin_amdgcn_wmma_f32_16x16x32_bf16(false, A0.v, false, Bop0,
                                                        (short)0, z, false, false);
            z = __builtin_amdgcn_wmma_f32_16x16x32_bf16(false, A1.v, false, Bop1,
                                                        (short)0, z, false, false);

            // W = BOW * rcp(Z+eps); D layout == GEMM2 A layout -> pack in regs
#pragma unroll
            for (int r = 0; r < 8; ++r) {
                int   d  = d0 + lhalf * 8 + r;               // D row M
                float bw = BOW[(long)d * VV + v0 + lmod];    // D col N = lane%16
                float w  = bw * fastrcp(z[r] + 1e-6f);
                Wop[sub * 8 + r] = (bf16_t)w;
            }
        }

        // GEMM2: acc[v, k'] += W(16v x 32doc) @ aT-slice(32doc x 16k')
#pragma unroll
        for (int t = 0; t < 4; ++t) {
            // B operand: col n = k' = 16t+lane%16, docs lhalf*8+s and 16+lhalf*8+s from aT
            V16U B2u;
            B2u.h[0] = *(const v8bf*)(aT + (long)(16 * t + lmod) * BB + db * 32 + lhalf * 8);
            B2u.h[1] = *(const v8bf*)(aT + (long)(16 * t + lmod) * BB + db * 32 + 16 + lhalf * 8);
            const v16bf B2 = B2u.v;
            acc[t] = __builtin_amdgcn_wmma_f32_16x16x32_bf16(false, Wop, false, B2,
                                                             (short)0, acc[t],
                                                             false, false);
        }
    }

    // ---- epilogue: out = (1-rho)*exp_n + rho*scale * b * temp
    // D layout: k = 16t + lane%16, v = v0 + lhalf*8 + r
#pragma unroll
    for (int t = 0; t < 4; ++t) {
        int k = 16 * t + lmod;
        if (k < KK) {
            float rd = fastrcp(denom[k]);
#pragma unroll
            for (int r = 0; r < 8; ++r) {
                int  v   = v0 + lhalf * 8 + r;
                long idx = (long)v * KK + k;
                float en = expn[idx];
                float bb = (beta[idx] + en) * rd;
                out[idx] = (1.0f - rho) * en + rs * bb * acc[t][r];
            }
        }
    }
}

// ---------------------------------------------------------------------------
extern "C" void kernel_launch(void* const* d_in, const int* in_sizes, int n_in,
                              void* d_out, int out_size, void* d_ws, size_t ws_size,
                              hipStream_t stream) {
    const float*     BOW    = (const float*)d_in[0];
    const float*     alpha  = (const float*)d_in[1];
    const float*     beta   = (const float*)d_in[2];
    const float*     expm   = (const float*)d_in[3];
    const float*     expn   = (const float*)d_in[4];
    const int*       bidx   = (const int*)d_in[5];
    const int*       iter_n = (const int*)d_in[6];
    const long long* batchC = (const long long*)d_in[7];
    const long long* C_m    = (const long long*)d_in[8];
    float*           out    = (float*)d_out;

    char*   ws    = (char*)d_ws;
    float*  denom = (float*)ws;                         // 64 f32
    bf16_t* aR    = (bf16_t*)(ws + 256);                // 64 KB, 256B aligned
    bf16_t* aT    = (bf16_t*)(ws + 256 + 64 * 1024);    // 64 KB

    hipMemsetAsync(denom, 0, 64 * sizeof(float), stream);
    mixehr_denom_kernel<<<256, 256, 0, stream>>>(beta, expn, denom);
    mixehr_prep_a_kernel<<<128, 256, 0, stream>>>(alpha, expm, bidx, aR, aT);

    const int blocks = (NSTRIP + 7) / 8;                // 8 waves (strips) / block
    mixehr_main_kernel<<<blocks, 256, 0, stream>>>(BOW, beta, expn, denom, aR, aT,
                                                   iter_n, batchC, C_m, out);
}
